// MoE_21698174779633
// MI455X (gfx1250) — compile-verified, matrix operand order checked by reading, not run on hardware
//
#include <hip/hip_runtime.h>
#include <hip/hip_bf16.h>
#include <math.h>

// ---------------------------------------------------------------------------
// MoE fused kernel for gfx1250 (MI455X), wave32, WMMA bf16 split-precision.
//   N=8192 tokens, D=1024, H=2048, E=8 experts, top-2 gating, fp32 in/out.
// M=32 tokens per workgroup (2 row-tiles/wave) so each weight-fragment load
// feeds 6 WMMAs -> halves L2 weight traffic vs M=16. GEMM2 K is processed in
// two 1024-wide chunks through a shared 128KB LDS h-fragment buffer.
// ---------------------------------------------------------------------------

typedef __attribute__((ext_vector_type(16))) __bf16        v16bf;
typedef __attribute__((ext_vector_type(8)))  float         v8f;
typedef __attribute__((ext_vector_type(8)))  unsigned int  v8u;

#define N_TOK 8192
#define DDIM  1024
#define HDIM  2048
#define NEXP  8
#define LN_EPS 1e-5f

// round-to-nearest-even f32 -> bf16 bits
__device__ __forceinline__ unsigned short f2bf(float f) {
    unsigned int u = __builtin_bit_cast(unsigned int, f);
    unsigned int r = u + 0x7FFFu + ((u >> 16) & 1u);
    return (unsigned short)(r >> 16);
}
__device__ __forceinline__ float bf2f(unsigned short h) {
    unsigned int u = ((unsigned int)h) << 16;
    return __builtin_bit_cast(float, u);
}
__device__ __forceinline__ v8f vzero() {
    v8f z;
#pragma unroll
    for (int i = 0; i < 8; ++i) z[i] = 0.f;
    return z;
}
__device__ __forceinline__ v16bf ldfrag(const unsigned short* p) {
    return __builtin_bit_cast(v16bf, *(const v8u*)p);
}

// ---------------------------------------------------------------------------
// Preprocess: split fp32 weights [E, Kd, Nd] into hi/lo bf16 in WMMA
// B-fragment order: dst[e][kb=k/32][t=n/16][lane][j], 32B contiguous per lane.
// ---------------------------------------------------------------------------
__global__ void prep_split_kernel(const float* __restrict__ src,
                                  unsigned short* __restrict__ dh,
                                  unsigned short* __restrict__ dl,
                                  int Kd, int Nd, long total) {
    long stride = (long)gridDim.x * blockDim.x;
    for (long idx = (long)blockIdx.x * blockDim.x + threadIdx.x; idx < total;
         idx += stride) {
        long per = (long)Kd * Nd;
        int  e   = (int)(idx / per);
        long r   = idx - (long)e * per;
        int  k   = (int)(r / Nd);
        int  n   = (int)(r - (long)k * Nd);
        int  kb  = k >> 5, kk = k & 31;
        int  t16 = n >> 4, nn = n & 15;
        int  lane = nn + 16 * ((kk >> 3) & 1);
        int  j    = (kk & 7) + 8 * (kk >> 4);
        long dst  = ((((long)e * (Kd >> 5) + kb) * (Nd >> 4) + t16) * 32 + lane) * 16 + j;
        float f = src[idx];
        unsigned short hi = f2bf(f);
        unsigned short lo = f2bf(f - bf2f(hi));
        dh[dst] = hi;
        dl[dst] = lo;
    }
}

// ---------------------------------------------------------------------------
// Fused MoE: gate -> (per expert) GEMM1 -> LN -> GELU -> GEMM2 -> combine.
// 1 workgroup = 32 tokens x full H. 16 waves x 128 H-columns, 2 row-tiles.
// ---------------------------------------------------------------------------
__global__ __launch_bounds__(512, 1) void moe_fused_kernel(
    const float* __restrict__ x,  const float* __restrict__ b1,
    const float* __restrict__ lng, const float* __restrict__ lnb,
    const float* __restrict__ b2, const float* __restrict__ Wg,
    const float* __restrict__ bg,
    const unsigned short* __restrict__ w1h, const unsigned short* __restrict__ w1l,
    const unsigned short* __restrict__ w2h, const unsigned short* __restrict__ w2l,
    float* __restrict__ out) {
    extern __shared__ unsigned char smem_raw[];
    unsigned short* xh = (unsigned short*)smem_raw;  // [2 mt][32 kb][32 lane][16]
    unsigned short* xl = xh + 2 * 32 * 32 * 16;      // 32768 ush each
    unsigned short* hh = xl + 2 * 32 * 32 * 16;      // K-chunk: [2 mt][32 kb][32][16]
    unsigned short* hl = hh + 2 * 32 * 32 * 16;
    float* gatew = (float*)(hl + 2 * 32 * 32 * 16);  // [32 tok][8 exp]
    float* rsum  = gatew + 32 * 8;                   // [32]
    float* rsq   = rsum + 32;
    float* s_mu  = rsq + 32;
    float* s_rs  = s_mu + 32;

    const int tid  = threadIdx.x;
    const int lane = tid & 31;
    const int wave = tid >> 5;
    const int hlf  = (lane >> 4) & 1;   // 16-lane half
    const int ln16 = lane & 15;
    const int cw   = wave >> 3;         // K-chunk this wave's columns fall into
    const int tile_base = blockIdx.x * 32;

    // ---- Phase 0: stage x tile [32 x 1024] into A-fragment layout (hi/lo) ----
    for (int it = 0; it < 64; ++it) {
        int i = tid + 512 * it;            // 0..32767
        int m = i >> 10;                   // token row 0..31
        int k = i & 1023;
        float f = x[(long)(tile_base + m) * DDIM + k];
        unsigned short hi = f2bf(f);
        unsigned short lo = f2bf(f - bf2f(hi));
        int mt = m >> 4, mm = m & 15;
        int kb = k >> 5, kk = k & 31;
        int l2 = mm + 16 * ((kk >> 3) & 1);
        int j  = (kk & 7) + 8 * (kk >> 4);
        int dst = ((mt * 32 + kb) * 32 + l2) * 16 + j;
        xh[dst] = hi;
        xl[dst] = lo;
    }
    __syncthreads();

    // ---- Gating: logits, softmax, top-2, renormalize ----
    if (tid < 256) {
        int m = tid >> 3, e = tid & 7;
        float s = bg[e];
        const float* xr = x + (long)(tile_base + m) * DDIM;
        for (int k = 0; k < DDIM; ++k) s += xr[k] * Wg[k * NEXP + e];
        gatew[m * 8 + e] = s;
    }
    __syncthreads();
    if (tid < 32) {
        int m = tid;
        float l[8], mx = -1e30f;
#pragma unroll
        for (int e = 0; e < 8; ++e) { l[e] = gatew[m * 8 + e]; mx = fmaxf(mx, l[e]); }
        float p[8];
#pragma unroll
        for (int e = 0; e < 8; ++e) p[e] = expf(l[e] - mx);
        int i0 = 0;
#pragma unroll
        for (int e = 1; e < 8; ++e) if (p[e] > p[i0]) i0 = e;
        int i1 = (i0 == 0) ? 1 : 0;
#pragma unroll
        for (int e = 0; e < 8; ++e) if (e != i0 && p[e] > p[i1]) i1 = e;
        float inv = 1.f / (p[i0] + p[i1]);   // softmax denom cancels in renorm
#pragma unroll
        for (int e = 0; e < 8; ++e)
            gatew[m * 8 + e] = (e == i0) ? p[i0] * inv : (e == i1) ? p[i1] * inv : 0.f;
    }
    __syncthreads();

    // ---- Expert loop ----
    for (int e = 0; e < NEXP; ++e) {
        if (tid < 32) { rsum[tid] = 0.f; rsq[tid] = 0.f; }
        __syncthreads();  // A

        // GEMM1: h = x @ W1[e]   (K = 1024, 32 steps of 32)
        v8f acc[2][8];
#pragma unroll
        for (int mt = 0; mt < 2; ++mt)
#pragma unroll
            for (int t = 0; t < 8; ++t) acc[mt][t] = vzero();

        for (int kb = 0; kb < 32; ++kb) {
            v16bf ah[2], al[2];
#pragma unroll
            for (int mt = 0; mt < 2; ++mt) {
                ah[mt] = ldfrag(xh + ((mt * 32 + kb) * 32 + lane) * 16);
                al[mt] = ldfrag(xl + ((mt * 32 + kb) * 32 + lane) * 16);
            }
            long tb = ((long)e * 32 + kb) * 128 + wave * 8;   // B tile index
            const unsigned short* ph = w1h + (tb * 32 + lane) * 16;
            const unsigned short* pl = w1l + (tb * 32 + lane) * 16;
            __builtin_prefetch(ph + 128 * 512, 0, 0);          // next kb step
            __builtin_prefetch(pl + 128 * 512, 0, 0);
#pragma unroll
            for (int t = 0; t < 8; ++t) {
                v16bf bh = ldfrag(ph + t * 512);
                v16bf bl = ldfrag(pl + t * 512);
#pragma unroll
                for (int mt = 0; mt < 2; ++mt) {
                    acc[mt][t] = __builtin_amdgcn_wmma_f32_16x16x32_bf16(false, ah[mt], false, bh, (short)0, acc[mt][t], false, false);
                    acc[mt][t] = __builtin_amdgcn_wmma_f32_16x16x32_bf16(false, ah[mt], false, bl, (short)0, acc[mt][t], false, false);
                    acc[mt][t] = __builtin_amdgcn_wmma_f32_16x16x32_bf16(false, al[mt], false, bh, (short)0, acc[mt][t], false, false);
                }
            }
        }

        // +b1 and LayerNorm partial sums
        float gsum[2][8], gsq[2][8];
#pragma unroll
        for (int mt = 0; mt < 2; ++mt)
#pragma unroll
            for (int r = 0; r < 8; ++r) { gsum[mt][r] = 0.f; gsq[mt][r] = 0.f; }
#pragma unroll
        for (int t = 0; t < 8; ++t) {
            int col = wave * 128 + t * 16 + ln16;
            float bb = b1[e * HDIM + col];
#pragma unroll
            for (int mt = 0; mt < 2; ++mt)
#pragma unroll
                for (int r = 0; r < 8; ++r) {
                    float v = acc[mt][t][r] + bb;
                    acc[mt][t][r] = v;
                    gsum[mt][r] += v;
                    gsq[mt][r]  += v * v;
                }
        }
#pragma unroll
        for (int msk = 1; msk <= 8; msk <<= 1) {  // reduce across 16-lane half
#pragma unroll
            for (int mt = 0; mt < 2; ++mt)
#pragma unroll
                for (int r = 0; r < 8; ++r) {
                    gsum[mt][r] += __shfl_xor(gsum[mt][r], msk, 32);
                    gsq[mt][r]  += __shfl_xor(gsq[mt][r],  msk, 32);
                }
        }
        if (ln16 == 0) {
#pragma unroll
            for (int mt = 0; mt < 2; ++mt)
#pragma unroll
                for (int r = 0; r < 8; ++r) {
                    atomicAdd(&rsum[mt * 16 + 8 * hlf + r], gsum[mt][r]);
                    atomicAdd(&rsq [mt * 16 + 8 * hlf + r], gsq[mt][r]);
                }
        }
        __syncthreads();  // B
        if (tid < 32) {
            float mu  = rsum[tid] * (1.f / HDIM);
            float var = rsq[tid] * (1.f / HDIM) - mu * mu;
            s_mu[tid] = mu;
            s_rs[tid] = rsqrtf(var + LN_EPS);
        }
        __syncthreads();  // C

        // GEMM2 over two K-chunks of 1024, staged through shared h buffer
        v8f acc2[2][8];
#pragma unroll
        for (int mt = 0; mt < 2; ++mt)
#pragma unroll
            for (int t = 0; t < 8; ++t) acc2[mt][t] = vzero();

        for (int c = 0; c < 2; ++c) {
            if (cw == c) {  // this wave's columns belong to chunk c: emit h frags
#pragma unroll
                for (int t = 0; t < 8; ++t) {
                    int col = wave * 128 + t * 16 + ln16;
                    float g = lng[e * HDIM + col];
                    float b = lnb[e * HDIM + col];
                    int kb_local = (col >> 5) - c * 32;
                    int kk    = col & 31;
                    int j     = (kk & 7) + 8 * (kk >> 4);
                    int lhalf = (kk >> 3) & 1;
#pragma unroll
                    for (int mt = 0; mt < 2; ++mt)
#pragma unroll
                        for (int r = 0; r < 8; ++r) {
                            int rr = 8 * hlf + r;   // row within mtile (0..15)
                            int row = mt * 16 + rr;
                            float v = (acc[mt][t][r] - s_mu[row]) * s_rs[row];
                            v = v * g + b;
                            v = 0.5f * v * (1.f + erff(v * 0.70710678118654752f));
                            unsigned short hi = f2bf(v);
                            unsigned short lo = f2bf(v - bf2f(hi));
                            int dst = ((mt * 32 + kb_local) * 32 + (rr + 16 * lhalf)) * 16 + j;
                            hh[dst] = hi;
                            hl[dst] = lo;
                        }
                }
            }
            __syncthreads();  // h chunk ready

            for (int kb = 0; kb < 32; ++kb) {
                v16bf ah[2], al[2];
#pragma unroll
                for (int mt = 0; mt < 2; ++mt) {
                    ah[mt] = ldfrag(hh + ((mt * 32 + kb) * 32 + lane) * 16);
                    al[mt] = ldfrag(hl + ((mt * 32 + kb) * 32 + lane) * 16);
                }
                long tb = ((long)e * 64 + c * 32 + kb) * 128 + wave * 8;
                const unsigned short* ph = w2h + (tb * 32 + lane) * 16;
                const unsigned short* pl = w2l + (tb * 32 + lane) * 16;
                __builtin_prefetch(ph + 128 * 512, 0, 0);
                __builtin_prefetch(pl + 128 * 512, 0, 0);
#pragma unroll
                for (int t = 0; t < 8; ++t) {
                    v16bf bh = ldfrag(ph + t * 512);
                    v16bf bl = ldfrag(pl + t * 512);
#pragma unroll
                    for (int mt = 0; mt < 2; ++mt) {
                        acc2[mt][t] = __builtin_amdgcn_wmma_f32_16x16x32_bf16(false, ah[mt], false, bh, (short)0, acc2[mt][t], false, false);
                        acc2[mt][t] = __builtin_amdgcn_wmma_f32_16x16x32_bf16(false, ah[mt], false, bl, (short)0, acc2[mt][t], false, false);
                        acc2[mt][t] = __builtin_amdgcn_wmma_f32_16x16x32_bf16(false, al[mt], false, bh, (short)0, acc2[mt][t], false, false);
                    }
                }
            }
            __syncthreads();  // protect h buffer before next chunk's writes
        }

        // combine into output: single-owner tile -> plain RMW, no atomics
        float gw[2][8];
#pragma unroll
        for (int mt = 0; mt < 2; ++mt)
#pragma unroll
            for (int r = 0; r < 8; ++r)
                gw[mt][r] = gatew[(mt * 16 + 8 * hlf + r) * 8 + e];
#pragma unroll
        for (int t = 0; t < 8; ++t) {
            int col = wave * 128 + t * 16 + ln16;
            float bb = b2[e * HDIM + col];
#pragma unroll
            for (int mt = 0; mt < 2; ++mt)
#pragma unroll
                for (int r = 0; r < 8; ++r) {
                    int row = tile_base + mt * 16 + 8 * hlf + r;
                    long o = (long)row * HDIM + col;
                    float val = (acc2[mt][t][r] + bb) * gw[mt][r];
                    if (e == 0) out[o] = val;
                    else        out[o] += val;
                }
        }
    }
}

// ---------------------------------------------------------------------------
extern "C" void kernel_launch(void* const* d_in, const int* in_sizes, int n_in,
                              void* d_out, int out_size, void* d_ws, size_t ws_size,
                              hipStream_t stream) {
    const float* x   = (const float*)d_in[0];
    const float* W1  = (const float*)d_in[1];
    const float* b1  = (const float*)d_in[2];
    const float* lng = (const float*)d_in[3];
    const float* lnb = (const float*)d_in[4];
    const float* W2  = (const float*)d_in[5];
    const float* b2  = (const float*)d_in[6];
    const float* Wg  = (const float*)d_in[7];
    const float* bg  = (const float*)d_in[8];
    float* out = (float*)d_out;

    const size_t n_w1 = (size_t)NEXP * DDIM * HDIM;   // 16,777,216
    const size_t n_w2 = (size_t)NEXP * HDIM * HDIM;   // 33,554,432
    unsigned short* w1h = (unsigned short*)d_ws;
    unsigned short* w1l = w1h + n_w1;
    unsigned short* w2h = w1l + n_w1;
    unsigned short* w2l = w2h + n_w2;

    prep_split_kernel<<<4096, 256, 0, stream>>>(W1, w1h, w1l, DDIM, HDIM, (long)n_w1);
    prep_split_kernel<<<8192, 256, 0, stream>>>(W2, w2h, w2l, HDIM, HDIM, (long)n_w2);

    // dynamic LDS: x frags 128KB + h-chunk frags 128KB + gate/LN scratch
    size_t lds = (size_t)(4 * 2 * 32 * 32 * 16) * sizeof(unsigned short)
               + (size_t)(32 * 8 + 4 * 32) * sizeof(float);
    moe_fused_kernel<<<N_TOK / 32, 512, lds, stream>>>(x, b1, lng, lnb, b2, Wg, bg,
                                                       w1h, w1l, w2h, w2l, out);
}
